// ImprovedModel_86638080295045
// MI455X (gfx1250) — matrix operand order repaired and checked by reading, not compiled, not run on hardware
//
#include <hip/hip_runtime.h>
#include <cstdint>
#include <cstddef>

// ---------------------------------------------------------------------------
// ImprovedModel pipeline for MI455X (gfx1250), wave32 + WMMA bf16 + TDM.
//   1) gather pert embedding -> combined[:, :128]
//   2) h        = relu(baseline @ ge_W1^T + b)          (WMMA, W-tile via TDM)
//   3) gene_emb = relu(h @ ge_W2^T + b) -> combined[:,128:]
//   4) g        = relu(combined @ grn_W1^T + b)
//   5) A_bf16   = bf16(0.5*tanh(g @ grn_W2^T + b))      (dominant 134MB stream)
//   6) y: 24 fixed-point iterations y = (x + a*A^T y)/(1+r); A bf16, L2-resident
//   7) d        = relu(y @ dec_W1^T + b)
//   8) out      = baseline + d @ dec_W2^T + b
// ---------------------------------------------------------------------------

typedef __attribute__((ext_vector_type(16))) __bf16 v16bf;
typedef __attribute__((ext_vector_type(8)))  float  v8f;
typedef __attribute__((ext_vector_type(4)))  unsigned int tdm_g0_t;
typedef __attribute__((ext_vector_type(8)))  int          tdm_g1_t;
typedef __attribute__((ext_vector_type(4)))  int          tdm_g23_t;

#define BATCH 128
#define GDIM  512
#define HDIM  128
#define GG    (GDIM * GDIM)
#define SOLVE_ITERS 24

__device__ __forceinline__ float bf2f(unsigned short u) {
  union { unsigned int ui; float f; } v;
  v.ui = ((unsigned int)u) << 16;
  return v.f;
}

__device__ __forceinline__ unsigned short f2bf_rne(float f) {
  union { float f; unsigned int u; } v;
  v.f = f;
  unsigned int r = v.u + 0x7FFFu + ((v.u >> 16) & 1u);
  return (unsigned short)(r >> 16);
}

// ---------------------------------------------------------------------------
// Stage a contiguous `bytes/4`-float run from global into LDS.
// Preferred path: Tensor Data Mover (1-D tile descriptor, ISA ch.8 layout).
// Issued by the calling wave only; caller must __syncthreads() afterwards.
// ---------------------------------------------------------------------------
__device__ __forceinline__ void tdm_copy_to_lds(const float* src, float* lds_dst,
                                                unsigned int n_elems) {
#if __has_builtin(__builtin_amdgcn_tensor_load_to_lds)
  const unsigned long long gaddr = (unsigned long long)(const void*)src;
  const unsigned int lds_addr = (unsigned int)(unsigned long long)(void*)lds_dst;

  tdm_g0_t g0;
  g0[0] = 1u;                                   // count=1, no gather, no restore
  g0[1] = lds_addr;                             // lds_addr [63:32]
  g0[2] = (unsigned int)gaddr;                  // global_addr low
  g0[3] = (unsigned int)((gaddr >> 32) & 0x01FFFFFFu) | (2u << 30); // type=2

  tdm_g1_t g1;
  g1[0] = (int)(2u << 16);                      // wg_mask=0, data_size=2 (4B)
  g1[1] = (int)((n_elems & 0xFFFFu) << 16);     // tensor_dim0[15:0] @ bits 63:48
  g1[2] = (int)((n_elems >> 16) | (1u << 16));  // tensor_dim0[31:16]; tensor_dim1=1
  g1[3] = (int)((n_elems & 0xFFFFu) << 16);     // tile_dim0 @ bits 127:112
  g1[4] = 1;                                    // tile_dim1=1, tile_dim2=0
  g1[5] = (int)n_elems;                         // tensor_dim0_stride low 32
  g1[6] = 0;
  g1[7] = 0;

  tdm_g23_t z4; z4[0] = 0; z4[1] = 0; z4[2] = 0; z4[3] = 0;
#if defined(__clang_major__) && __clang_major__ >= 23
  tdm_g1_t z8; z8[0]=0; z8[1]=0; z8[2]=0; z8[3]=0; z8[4]=0; z8[5]=0; z8[6]=0; z8[7]=0;
  __builtin_amdgcn_tensor_load_to_lds(g0, g1, z4, z4, z8, 0);
#else
  __builtin_amdgcn_tensor_load_to_lds(g0, g1, z4, z4, 0);
#endif
#if __has_builtin(__builtin_amdgcn_s_wait_tensorcnt)
  __builtin_amdgcn_s_wait_tensorcnt(0);
#endif
#else
  // Fallback: this wave copies with float4s.
  const int lane = threadIdx.x & 31;
  const float4* s4 = (const float4*)src;
  float4* d4 = (float4*)lds_dst;
  for (unsigned int i = lane; i < n_elems / 4; i += 32) d4[i] = s4[i];
#endif
}

// ---------------------------------------------------------------------------
// Generic WMMA GEMM: C(M=128 x N) = epi(A(128 x KV) @ W(N x KV)^T + bias)
// W must be contiguous row-major with ldw == KV (true for all layers here).
// One block = 256 threads = 8 wave32s; wave w owns m-tile [16w, 16w+16);
// grid.x = N/16. W-tile (16 x KV fp32) staged once into LDS via TDM.
// EPI: 0 = relu -> f32, 1 = 0.5*tanh -> bf16 (row stride N), 2 = +resid -> f32
// ---------------------------------------------------------------------------
template <int EPI, int KV>
__global__ __launch_bounds__(256) void wmma_gemm_kernel(
    const float* __restrict__ A, int lda,
    const float* __restrict__ W,
    const float* __restrict__ bias,
    float* __restrict__ Cf, int ldc,
    unsigned short* __restrict__ Cbf,
    const float* __restrict__ resid, int ldr,
    int N) {
  __shared__ float wtile[16 * KV];

  const int wave = threadIdx.x >> 5;
  const int lane = threadIdx.x & 31;
  const int half = lane >> 4;   // which 16-lane half of the wave
  const int l16  = lane & 15;
  const int mbase = wave << 4;
  const int nbase = blockIdx.x << 4;

  // Stage this block's 16 x KV weight tile (contiguous 64*KV bytes) into LDS.
  if (wave == 0) {
    tdm_copy_to_lds(W + (size_t)nbase * KV, wtile, 16u * (unsigned)KV);
  }
  if (nbase + 16 < N) {
    __builtin_prefetch(W + (size_t)(nbase + 16) * KV, 0, 1);  // next tile -> L2
  }
  __syncthreads();

  const float* __restrict__ Arow  = A + (size_t)(mbase + l16) * lda;
  const float* __restrict__ Wlds  = wtile + (size_t)l16 * KV;

  v8f acc = {};
  for (int kb = 0; kb < KV; kb += 32) {
    // A fragment (16x32 bf16): half 0 -> K {0..15}, half 1 -> {16..31}
    v16bf afrag;
    const int ka0 = kb + half * 16;
#pragma unroll
    for (int e = 0; e < 8; ++e) afrag[e]     = (__bf16)Arow[ka0 + e];
#pragma unroll
    for (int e = 0; e < 8; ++e) afrag[8 + e] = (__bf16)Arow[ka0 + 8 + e];

    // B fragment (32x16 bf16, B = W^T): 16 contiguous K from LDS (ds_load_b128 x4)
    v16bf bfrag;
    const int kb0 = kb + half * 16;
#pragma unroll
    for (int e = 0; e < 16; ++e) bfrag[e] = (__bf16)Wlds[kb0 + e];

    acc = __builtin_amdgcn_wmma_f32_16x16x32_bf16(
        false, afrag, false, bfrag, (short)0, acc, false, false);
  }

  // D layout: VGPR v, lane = 16*half + n16 -> element (m = 8*half + v, n = n16)
  const int n = nbase + l16;
  const float bn = bias ? bias[n] : 0.0f;
#pragma unroll
  for (int v = 0; v < 8; ++v) {
    const int m = mbase + half * 8 + v;
    float val = acc[v] + bn;
    if constexpr (EPI == 0) {
      Cf[(size_t)m * ldc + n] = fmaxf(val, 0.0f);
    } else if constexpr (EPI == 1) {
      Cbf[(size_t)m * (size_t)N + n] = f2bf_rne(0.5f * tanhf(val));
    } else {
      Cf[(size_t)m * ldc + n] = val + resid[(size_t)m * ldr + n];
    }
  }
}

// ---------------------------------------------------------------------------
// Perturbation embedding gather -> combined[:, 0:H]
// ---------------------------------------------------------------------------
__global__ __launch_bounds__(128) void gather_embed_kernel(
    const int* __restrict__ pert, const float* __restrict__ embW,
    float* __restrict__ combined) {
  const int m = blockIdx.x;
  const int t = threadIdx.x;
  combined[(size_t)m * (2 * HDIM) + t] = embW[(size_t)pert[m] * HDIM + t];
}

// ---------------------------------------------------------------------------
// Solve ((1+r)I - a*A^T) y = x per sample via fixed-point iteration:
//   y <- x/(1+r) + (a/(1+r)) * A^T y
// A^T y as row-accumulation over row-major bf16 A. 512 threads/block:
// thread t owns columns 4*(t&127)..+3 over row-quarter (t>>7); partials
// reduced in LDS. 128 blocks x 16 waves = 2048 waves of L2-resident streaming.
// ---------------------------------------------------------------------------
__global__ __launch_bounds__(512) void grn_solve_kernel(
    const unsigned short* __restrict__ Abf,
    const float* __restrict__ xin, float* __restrict__ yout) {
  const int b = blockIdx.x;
  const int t = threadIdx.x;            // 0..511
  const int c0 = (t & 127) * 4;         // first owned column
  const int q  = t >> 7;                // row quarter (0..3)
  __shared__ float ylds[GDIM];
  __shared__ float part[4][GDIM];

  const unsigned short* __restrict__ Ab = Abf + (size_t)b * GG;
  const float* __restrict__ xb = xin + (size_t)b * GDIM;

  const float inv = 1.0f / (1.0f + 1e-6f);
  const float cf  = 0.1f * inv;

  const float xt = xb[t] * inv;         // this thread's reduction column
  ylds[t] = xt;
  __syncthreads();

  const ushort4* __restrict__ Arow4 =
      (const ushort4*)(Ab + (size_t)(q * 128) * GDIM + c0);
  for (int it = 0; it < SOLVE_ITERS; ++it) {
    float a0 = 0.f, a1 = 0.f, a2 = 0.f, a3 = 0.f;
    const float* yq = ylds + q * 128;
#pragma unroll 4
    for (int j = 0; j < 128; ++j) {
      const float yj = yq[j];                          // LDS broadcast
      const ushort4 w = Arow4[(size_t)j * (GDIM / 4)]; // 8B coalesced per lane
      a0 = fmaf(yj, bf2f(w.x), a0);
      a1 = fmaf(yj, bf2f(w.y), a1);
      a2 = fmaf(yj, bf2f(w.z), a2);
      a3 = fmaf(yj, bf2f(w.w), a3);
    }
    part[q][c0 + 0] = a0;
    part[q][c0 + 1] = a1;
    part[q][c0 + 2] = a2;
    part[q][c0 + 3] = a3;
    __syncthreads();
    // thread t reduces column t across the 4 row-quarters
    const float s = part[0][t] + part[1][t] + part[2][t] + part[3][t];
    __syncthreads();
    ylds[t] = xt + cf * s;
    __syncthreads();
  }

  yout[(size_t)b * GDIM + t] = ylds[t];
}

// ---------------------------------------------------------------------------
extern "C" void kernel_launch(void* const* d_in, const int* in_sizes, int n_in,
                              void* d_out, int out_size, void* d_ws, size_t ws_size,
                              hipStream_t stream) {
  (void)in_sizes; (void)n_in; (void)out_size; (void)ws_size;

  const int*   pert     = (const int*)d_in[0];
  const float* baseline = (const float*)d_in[1];   // (128, 512)
  const float* embW     = (const float*)d_in[2];   // (1000, 128)
  const float* geW1     = (const float*)d_in[3];   // (128, 512)
  const float* geB1     = (const float*)d_in[4];
  const float* geW2     = (const float*)d_in[5];   // (128, 128)
  const float* geB2     = (const float*)d_in[6];
  const float* grnW1    = (const float*)d_in[7];   // (128, 256)
  const float* grnB1    = (const float*)d_in[8];
  const float* grnW2    = (const float*)d_in[9];   // (262144, 128)
  const float* grnB2    = (const float*)d_in[10];  // (262144,)
  const float* decW1    = (const float*)d_in[11];  // (128, 512)
  const float* decB1    = (const float*)d_in[12];
  const float* decW2    = (const float*)d_in[13];  // (512, 128)
  const float* decB2    = (const float*)d_in[14];
  float* out = (float*)d_out;                      // (128, 512)

  char* ws = (char*)d_ws;
  float*          combined = (float*)(ws + 0);          // 128 x 256
  float*          hbuf     = (float*)(ws + 131072);     // 128 x 128
  float*          gbuf     = (float*)(ws + 196608);     // 128 x 128
  float*          ybuf     = (float*)(ws + 262144);     // 128 x 512
  float*          dbuf     = (float*)(ws + 524288);     // 128 x 128
  unsigned short* Abf      = (unsigned short*)(ws + 589824); // 128 x 512 x 512 bf16

  // 1) pert embedding -> combined[:, :128]
  gather_embed_kernel<<<BATCH, HDIM, 0, stream>>>(pert, embW, combined);

  // 2) h = relu(baseline @ geW1^T + b)
  wmma_gemm_kernel<0, GDIM><<<HDIM / 16, 256, 0, stream>>>(
      baseline, GDIM, geW1, geB1, hbuf, HDIM, nullptr, nullptr, 0, HDIM);

  // 3) gene_emb = relu(h @ geW2^T + b) -> combined[:, 128:]
  wmma_gemm_kernel<0, HDIM><<<HDIM / 16, 256, 0, stream>>>(
      hbuf, HDIM, geW2, geB2, combined + HDIM, 2 * HDIM, nullptr, nullptr, 0, HDIM);

  // 4) g = relu(combined @ grnW1^T + b)
  wmma_gemm_kernel<0, 2 * HDIM><<<HDIM / 16, 256, 0, stream>>>(
      combined, 2 * HDIM, grnW1, grnB1, gbuf, HDIM, nullptr, nullptr, 0, HDIM);

  // 5) A = bf16(0.5 * tanh(g @ grnW2^T + b))   [dominant 134MB+67MB stream]
  wmma_gemm_kernel<1, HDIM><<<GG / 16, 256, 0, stream>>>(
      gbuf, HDIM, grnW2, grnB2, nullptr, 0, Abf, nullptr, 0, GG);

  // 6) iterative solve ((1+r)I - a*A^T) y = baseline
  grn_solve_kernel<<<BATCH, GDIM, 0, stream>>>(Abf, baseline, ybuf);

  // 7) d = relu(y @ decW1^T + b)
  wmma_gemm_kernel<0, GDIM><<<HDIM / 16, 256, 0, stream>>>(
      ybuf, GDIM, decW1, decB1, dbuf, HDIM, nullptr, nullptr, 0, HDIM);

  // 8) out = baseline + d @ decW2^T + b
  wmma_gemm_kernel<2, HDIM><<<GDIM / 16, 256, 0, stream>>>(
      dbuf, HDIM, decW2, decB2, out, GDIM, nullptr, baseline, GDIM, GDIM);
}
